// RelativeAttentionHead_67989332296029
// MI455X (gfx1250) — matrix operand order repaired and checked by reading, not compiled, not run on hardware
//
#include <hip/hip_runtime.h>

typedef __attribute__((ext_vector_type(16))) _Float16 v16h;
typedef __attribute__((ext_vector_type(8)))  _Float16 v8h;
typedef __attribute__((ext_vector_type(8)))  float    v8f;

#define WMMA(a, b, c) __builtin_amdgcn_wmma_f32_16x16x32_f16( \
    false, (a), false, (b), (short)0, (c), false, false)

static __device__ inline v16h combine8(v8h lo, v8h hi) {
  return __builtin_shufflevector(lo, hi, 0,1,2,3,4,5,6,7,8,9,10,11,12,13,14,15);
}

// A-fragment (16x32 f16) from row-major f16 src (leading dim `ld` halves).
// ISA layout: lane = 16h+m holds row m, K = 8h+[0..7] (elems 0-7) and 16+8h+[0..7] (elems 8-15).
static __device__ inline v16h load_a16(const _Float16* src, int ld) {
  int lane = threadIdx.x & 31;
  int m = lane & 15, hh = lane >> 4;
  const _Float16* p = src + (size_t)m * ld + 8 * hh;
  v8h lo = *(const v8h*)p;
  v8h hi = *(const v8h*)(p + 16);
  return combine8(lo, hi);
}

// B-fragment (32x16 f16): B[k][n] = src[n*ld + k].
// ISA layout: lane = 16h+n holds col n, K = 8h+[0..7] (elems 0-7) and 16+8h+[0..7] (elems 8-15).
static __device__ inline v16h load_b16(const _Float16* src, int ld) {
  int lane = threadIdx.x & 31;
  int n = lane & 15, hh = lane >> 4;
  const _Float16* p = src + (size_t)n * ld + 8 * hh;
  v8h lo = *(const v8h*)p;
  v8h hi = *(const v8h*)(p + 16);
  return combine8(lo, hi);
}

// A-fragment from row-major f32 source, converting to f16.
static __device__ inline v16h load_a16_f32(const float* src, int ld) {
  int lane = threadIdx.x & 31;
  int m = lane & 15, hh = lane >> 4;
  const float* p = src + (size_t)m * ld + 8 * hh;
  float4 f0 = ((const float4*)p)[0];
  float4 f1 = ((const float4*)p)[1];
  float4 f2 = ((const float4*)(p + 16))[0];
  float4 f3 = ((const float4*)(p + 16))[1];
  v16h r;
  r[0]=(_Float16)f0.x;  r[1]=(_Float16)f0.y;  r[2]=(_Float16)f0.z;  r[3]=(_Float16)f0.w;
  r[4]=(_Float16)f1.x;  r[5]=(_Float16)f1.y;  r[6]=(_Float16)f1.z;  r[7]=(_Float16)f1.w;
  r[8]=(_Float16)f2.x;  r[9]=(_Float16)f2.y;  r[10]=(_Float16)f2.z; r[11]=(_Float16)f2.w;
  r[12]=(_Float16)f3.x; r[13]=(_Float16)f3.y; r[14]=(_Float16)f3.z; r[15]=(_Float16)f3.w;
  return r;
}

// B-fragment from row-major f32 source (B[k][n] = src[n*ld + k]), converting to f16.
static __device__ inline v16h load_b16_f32(const float* src, int ld) {
  int lane = threadIdx.x & 31;
  int n = lane & 15, hh = lane >> 4;
  const float* p = src + (size_t)n * ld + 8 * hh;
  float4 f0 = ((const float4*)p)[0];
  float4 f1 = ((const float4*)p)[1];
  float4 f2 = ((const float4*)(p + 16))[0];
  float4 f3 = ((const float4*)(p + 16))[1];
  v16h r;
  r[0]=(_Float16)f0.x;  r[1]=(_Float16)f0.y;  r[2]=(_Float16)f0.z;  r[3]=(_Float16)f0.w;
  r[4]=(_Float16)f1.x;  r[5]=(_Float16)f1.y;  r[6]=(_Float16)f1.z;  r[7]=(_Float16)f1.w;
  r[8]=(_Float16)f2.x;  r[9]=(_Float16)f2.y;  r[10]=(_Float16)f2.z; r[11]=(_Float16)f2.w;
  r[12]=(_Float16)f3.x; r[13]=(_Float16)f3.y; r[14]=(_Float16)f3.z; r[15]=(_Float16)f3.w;
  return r;
}

// ------------------------------------------------------------------
// TDM (Tensor Data Mover) tile load: descriptor per CDNA5 ISA.
// This toolchain exposes the 6-arg builtin: (g0, g1, g2, g3, g4, cpol).
// ------------------------------------------------------------------
#if __has_builtin(__builtin_amdgcn_tensor_load_to_lds)
#define HAVE_TDM 1
typedef unsigned int v4u __attribute__((ext_vector_type(4)));
typedef int v8i_ __attribute__((ext_vector_type(8)));
typedef int v4i_ __attribute__((ext_vector_type(4)));

// 2-D tile load (8-byte units). tile_d1 == 0 -> 1-D contiguous copy.
static __device__ inline void tdm_load_tile(unsigned lds_off, const void* gaddr,
                                            unsigned tile_d0_u, unsigned tile_d1,
                                            unsigned stride_u) {
  unsigned long long ga = (unsigned long long)(size_t)gaddr;
  unsigned td1 = tile_d1 ? tile_d1 : 1u;
  v4u g0;
  g0[0] = 1u;                                             // count=1, user mode
  g0[1] = lds_off;                                        // lds_addr (bytes)
  g0[2] = (unsigned)(ga & 0xFFFFFFFFu);                   // global_addr[31:0]
  g0[3] = (unsigned)((ga >> 32) & 0x1FFFFFFu) | (2u << 30); // addr[56:32] | type=2
  v8i_ g1;
  g1[0] = (int)(3u << 16);                                // data_size = 8B, no mask/flags
  g1[1] = (int)((tile_d0_u & 0xFFFFu) << 16);             // tensor_dim0[15:0] @ bits63:48
  g1[2] = (int)((tile_d0_u >> 16) | ((td1 & 0xFFFFu) << 16)); // dim0[31:16] | dim1[15:0]
  g1[3] = (int)((td1 >> 16) | ((tile_d0_u & 0xFFFFu) << 16)); // dim1[31:16] | tile_dim0
  g1[4] = (int)(tile_d1 & 0xFFFFu);                       // tile_dim1 (0 => 1-D)
  g1[5] = (int)stride_u;                                  // tensor_dim0_stride[31:0]
  g1[6] = 0;                                              // stride[47:32] | dim1_stride lo
  g1[7] = 0;
  v4i_ z4 = {0, 0, 0, 0};
  v8i_ z8 = {0, 0, 0, 0, 0, 0, 0, 0};
  __builtin_amdgcn_tensor_load_to_lds(g0, g1, z4, z4, z8, 0);
}

static __device__ inline void tdm_wait0() {
#if __has_builtin(__builtin_amdgcn_s_wait_tensorcnt)
  __builtin_amdgcn_s_wait_tensorcnt(0);
#else
  asm volatile("s_wait_tensorcnt 0x0" ::: "memory");
#endif
}
#else
#define HAVE_TDM 0
#endif

// ------------------------------------------------------------------
// Kernel C: convert rel_emb (2L-1, 64) f32 -> f16 workspace
// ------------------------------------------------------------------
__global__ void cvt_kernel(const float* __restrict__ src, _Float16* __restrict__ dst, int n) {
  int i = blockIdx.x * blockDim.x + threadIdx.x;
  if (i < n) dst[i] = (_Float16)src[i];
}

// ------------------------------------------------------------------
// Kernel A: fused QKV projection. Q,K stored (B*L, 64) f16 row-major,
// V stored transposed (B, 64, L) f16.
// ------------------------------------------------------------------
__global__ __launch_bounds__(256) void qkv_kernel(
    const float* __restrict__ x,
    const float* __restrict__ Wq, const float* __restrict__ Wk, const float* __restrict__ Wv,
    _Float16* __restrict__ Qo, _Float16* __restrict__ Ko, _Float16* __restrict__ Vto) {
  const int E = 512, H = 64, L = 2048;
  int wave = threadIdx.x >> 5, lane = threadIdx.x & 31;
  int m = lane & 15, hh = lane >> 4;
  int row0 = blockIdx.x * 128 + wave * 16;   // 16 output rows per wave

  v8f acc[3][4] = {};  // [Q/K/V][N-subtile]
  const float* Ws[3] = {Wq, Wk, Wv};

  for (int kk = 0; kk < E; kk += 32) {
    v16h a = load_a16_f32(x + (size_t)row0 * E + kk, E);
    #pragma unroll
    for (int mat = 0; mat < 3; ++mat) {
      #pragma unroll
      for (int nt = 0; nt < 4; ++nt) {
        v16h bfr = load_b16_f32(Ws[mat] + (size_t)(nt * 16) * E + kk, E);
        acc[mat][nt] = WMMA(a, bfr, acc[mat][nt]);
      }
    }
  }

  int b = row0 >> 11;        // row0 / 2048 (block never straddles a batch)
  int posb = row0 & 2047;
  #pragma unroll
  for (int nt = 0; nt < 4; ++nt) {
    #pragma unroll
    for (int e = 0; e < 8; ++e) {
      int r = e + 8 * hh;               // C layout: row = e (+8 for hi half)
      int col = nt * 16 + m;            // col = lane
      size_t grow = (size_t)(row0 + r);
      Qo[grow * H + col] = (_Float16)acc[0][nt][e];
      Ko[grow * H + col] = (_Float16)acc[1][nt][e];
      Vto[((size_t)b * H + col) * L + (size_t)(posb + r)] = (_Float16)acc[2][nt][e];
    }
  }
}

// ------------------------------------------------------------------
// Kernel B: causal flash attention with relative-position skew.
// ------------------------------------------------------------------
#define BM 128
#define BN 128

static __device__ inline v8f rel_tile(v16h qa0, v16h qa1, const _Float16* relb, int r0) {
  v8f c = {};
  v16h b0 = load_b16(relb + (size_t)r0 * 64, 64);
  c = WMMA(qa0, b0, c);
  v16h b1 = load_b16(relb + (size_t)r0 * 64 + 32, 64);
  c = WMMA(qa1, b1, c);
  return c;
}

// S[e] = S[e]*scale + M[row][15 - row + col]  (skewed rel add via lane shuffles)
static __device__ inline void add_skew(v8f& S, const v8f& M0, const v8f& M1, float scale) {
  int lane = threadIdx.x & 31;
  int m = lane & 15, hh = lane >> 4;
  #pragma unroll
  for (int e = 0; e < 8; ++e) {
    int row = e + 8 * hh;
    int c = 15 - row + m;                 // in [0,30]
    int src = (lane & 16) | (c & 15);
    float va = __shfl(M0[e], src, 32);
    float vb = __shfl(M1[e], src, 32);
    S[e] = S[e] * scale + ((c < 16) ? va : vb);
  }
}

__global__ __launch_bounds__(256) void attn_kernel(
    const _Float16* __restrict__ Qg, const _Float16* __restrict__ Kg,
    const _Float16* __restrict__ Vt, const _Float16* __restrict__ relb,
    float* __restrict__ out) {
  const int L = 2048, H = 64;
  __shared__ _Float16 sK[BN * 64];       // (key, h)
  __shared__ _Float16 sV[64 * BN];       // (h, key)  -- V transposed
  __shared__ _Float16 sP[8 * 16 * 32];   // per-wave P relayout scratch

  int b  = blockIdx.x / (L / BM);
  int qt = blockIdx.x % (L / BM);
  int wave = threadIdx.x >> 5, lane = threadIdx.x & 31;
  int m = lane & 15, hh = lane >> 4;
  int qw = qt * BM + wave * 16;          // wave's first query (within batch)

  const _Float16* Qb = Qg + (size_t)b * L * H;
  const _Float16* Kb = Kg + (size_t)b * L * H;
  const _Float16* Vb = Vt + (size_t)b * H * L;

  v16h qa0 = load_a16(Qb + (size_t)qw * H, H);        // Q k=0..31
  v16h qa1 = load_a16(Qb + (size_t)qw * H + 32, H);   // Q k=32..63

  v8f O[4] = {};
  float mrow[8], lrow[8];
  #pragma unroll
  for (int e = 0; e < 8; ++e) { mrow[e] = -1e30f; lrow[e] = 0.0f; }

  // rolling rel tile: tile(rb(js)) where rb(js) = L-16-qw+js; start at js=0
  v8f Mprev = rel_tile(qa0, qa1, relb, L - 16 - qw);

  const float inv_s = 0.125f;            // 1/sqrt(64)
  int nkt = qt + 1;                      // causal: key tiles 0..qt

  for (int jt = 0; jt < nkt; ++jt) {
    int j0 = jt * BN;
    __syncthreads();
#if HAVE_TDM
    // One wave drives the Tensor Data Mover; DMA goes straight to LDS.
    if (wave == 0) {
      // K tile: contiguous 16KB, 1-D (2048 x 8B units)
      tdm_load_tile((unsigned)(size_t)&sK[0], Kb + (size_t)j0 * 64,
                    2048u, 0u, 2048u);
      // V tile: 64 rows x 256B, global row stride L*2 bytes = 512 x 8B units
      tdm_load_tile((unsigned)(size_t)&sV[0], Vb + (size_t)j0,
                    32u, 64u, 512u);
      tdm_wait0();
    }
    if (jt + 1 < nkt) {
      __builtin_prefetch(Kb + (size_t)(j0 + BN) * 64 + threadIdx.x * 32, 0, 0);
    }
#else
    // cooperative fill of sK (flat 16KB, contiguous in Kb) and sV (64 rows of 128 keys)
    {
      const uint4* gk = (const uint4*)(Kb + (size_t)j0 * 64);
      uint4* lk = (uint4*)sK;
      uint4* lv = (uint4*)sV;
      #pragma unroll
      for (int i = 0; i < 4; ++i) {
        int idx = threadIdx.x + 256 * i;         // 0..1023
        lk[idx] = gk[idx];
        int h8 = idx >> 4;                        // V head row 0..63
        int seg = idx & 15;                       // 16B segment 0..15
        lv[h8 * 16 + seg] = *(const uint4*)(Vb + (size_t)h8 * L + j0 + seg * 8);
      }
      if (jt + 1 < nkt) {
        __builtin_prefetch(Kb + (size_t)(j0 + BN) * 64 + threadIdx.x * 32, 0, 0);
      }
    }
#endif
    __syncthreads();

    #pragma unroll 1
    for (int p4 = 0; p4 < 4; ++p4) {
      int js0 = j0 + p4 * 32;                     // first subtile key start
      if (js0 > qw + 15) break;                   // fully masked from here on
      bool do2 = (js0 + 16) <= (qw + 15);

      // S = Q K^T for two 16-key subtiles
      v8f S0 = {}, S1 = {};
      {
        v16h kb0 = load_b16(sK + (size_t)(p4 * 32) * 64, 64);
        v16h kb1 = load_b16(sK + (size_t)(p4 * 32) * 64 + 32, 64);
        S0 = WMMA(qa0, kb0, S0);
        S0 = WMMA(qa1, kb1, S0);
      }
      if (do2) {
        v16h kb0 = load_b16(sK + (size_t)(p4 * 32 + 16) * 64, 64);
        v16h kb1 = load_b16(sK + (size_t)(p4 * 32 + 16) * 64 + 32, 64);
        S1 = WMMA(qa0, kb0, S1);
        S1 = WMMA(qa1, kb1, S1);
      }

      // relative-position skew (rolling M tiles)
      int rb = L - 16 - qw + js0;
      v8f Mc = rel_tile(qa0, qa1, relb, rb + 16);
      add_skew(S0, Mprev, Mc, inv_s);
      Mprev = Mc;
      if (do2) {
        v8f Mc2 = rel_tile(qa0, qa1, relb, rb + 32);
        add_skew(S1, Mprev, Mc2, inv_s);
        Mprev = Mc2;
      }

      // causal mask
      #pragma unroll
      for (int e = 0; e < 8; ++e) {
        int qi = qw + e + 8 * hh;
        if (js0 + m > qi) S0[e] = -1e30f;
        if (!do2 || (js0 + 16 + m > qi)) S1[e] = -1e30f;
      }

      // online softmax (row stats via 16-lane butterfly in each half-wave)
      float P0[8], P1[8];
      #pragma unroll
      for (int e = 0; e < 8; ++e) {
        float t = fmaxf(S0[e], S1[e]);
        t = fmaxf(t, __shfl_xor(t, 1, 32));
        t = fmaxf(t, __shfl_xor(t, 2, 32));
        t = fmaxf(t, __shfl_xor(t, 4, 32));
        t = fmaxf(t, __shfl_xor(t, 8, 32));
        float mn = fmaxf(mrow[e], t);
        float sc = __expf(mrow[e] - mn);
        mrow[e] = mn;
        lrow[e] *= sc;
        O[0][e] *= sc; O[1][e] *= sc; O[2][e] *= sc; O[3][e] *= sc;
        float p0 = __expf(S0[e] - mn);
        float p1 = __expf(S1[e] - mn);
        P0[e] = p0; P1[e] = p1;
        float s = p0 + p1;
        s += __shfl_xor(s, 1, 32);
        s += __shfl_xor(s, 2, 32);
        s += __shfl_xor(s, 4, 32);
        s += __shfl_xor(s, 8, 32);
        lrow[e] += s;
      }

      // P (C layout) -> LDS -> A fragment
      _Float16* sp = sP + wave * (16 * 32);
      #pragma unroll
      for (int e = 0; e < 8; ++e) {
        int row = e + 8 * hh;
        sp[row * 32 + m]      = (_Float16)P0[e];
        sp[row * 32 + 16 + m] = (_Float16)P1[e];
      }
      v16h pa = load_a16(sp, 32);

      // O += P @ V  (B fragments contiguous thanks to transposed V)
      #pragma unroll
      for (int nt = 0; nt < 4; ++nt) {
        v16h vb = load_b16(sV + (size_t)(nt * 16) * BN + p4 * 32, BN);
        O[nt] = WMMA(pa, vb, O[nt]);
      }
    }
  }

  // write normalized output (f32)
  #pragma unroll
  for (int nt = 0; nt < 4; ++nt) {
    #pragma unroll
    for (int e = 0; e < 8; ++e) {
      int row = qw + e + 8 * hh;
      out[((size_t)b * L + row) * H + nt * 16 + m] = O[nt][e] / lrow[e];
    }
  }
}

// ------------------------------------------------------------------
extern "C" void kernel_launch(void* const* d_in, const int* in_sizes, int n_in,
                              void* d_out, int out_size, void* d_ws, size_t ws_size,
                              hipStream_t stream) {
  const float* x   = (const float*)d_in[0];
  const float* Wq  = (const float*)d_in[1];
  const float* Wk  = (const float*)d_in[2];
  const float* Wv  = (const float*)d_in[3];
  const float* rel = (const float*)d_in[4];
  float* out = (float*)d_out;

  const int B = 8, L = 2048, H = 64;
  const size_t NL = (size_t)B * L;       // 16384 rows

  _Float16* Qw = (_Float16*)d_ws;
  _Float16* Kw = Qw + NL * H;
  _Float16* Vw = Kw + NL * H;            // transposed (B, 64, L)
  _Float16* Rw = Vw + NL * H;            // (2L-1, 64)
  int nrel = (2 * L - 1) * H;            // 262080

  cvt_kernel<<<dim3((nrel + 255) / 256), dim3(256), 0, stream>>>(rel, Rw, nrel);
  qkv_kernel<<<dim3((int)(NL / 128)), dim3(256), 0, stream>>>(x, Wq, Wk, Wv, Qw, Kw, Vw);
  attn_kernel<<<dim3(B * (L / BM)), dim3(256), 0, stream>>>(Qw, Kw, Vw, Rw, out);
}
